// FourierGridModel_40948218200303
// MI455X (gfx1250) — matrix operand order repaired and checked by reading, not compiled, not run on hardware
//
#include <hip/hip_runtime.h>
#include <hip/hip_bf16.h>
#include <math.h>

typedef __attribute__((ext_vector_type(16))) _Float16 v16h;
typedef __attribute__((ext_vector_type(8)))  _Float16 v8h;
typedef __attribute__((ext_vector_type(8)))  float    v8f;

#define N_RAYS    4096
#define N_SAMPLES 192
#define GRID_D    160
#define K0_DIM    12
#define WIDTH     128
#define DIM0      39          // 12 + 3 + 12 + 12
#define KPAD      64
#define NEARP     0.2f
#define FARP      3.0f
#define BG        0.2f
#define XYZ_MIN   (-1.2f)
#define XYZ_MAX   (1.2f)
#define STEPDIST  ((FARP - NEARP) / (float)N_SAMPLES)
#define ACT_SHIFT (-4.59511985013459f)   // log(1/(1-0.01) - 1)
#define M_TOTAL   (N_RAYS * N_SAMPLES)   // 786432
#define ROWS_PER_BLOCK 128               // 8 waves x 16 rows

// ---------------------------------------------------------------------------
// Fragment builders for the CDNA5 wave32 WMMA register layouts (16-bit data).
// A (16x32, MxK): lane<16 -> M=lane,   elems 0..7 = K koff..koff+7,  8..15 = K koff+16..+23, koff=0
//                 lane>=16 -> M=lane-16, same with koff=8
// B (32x16, KxN): lane column N = lane&15; elems 0..7 = K ks..ks+7, 8..15 = K ks+16..+23, ks = (lane<16?0:8)
// C/D (16x16 f32): elem r -> row M = r + (lane<16?0:8), col N = lane&15
// ---------------------------------------------------------------------------
__device__ __forceinline__ v16h combine16(v8h lo, v8h hi) {
  v16h r;
#pragma unroll
  for (int i = 0; i < 8; ++i) { r[i] = lo[i]; r[i + 8] = hi[i]; }
  return r;
}

// row_base points at the first element of this lane's M row; kb in {0,32,64,96}
__device__ __forceinline__ v16h load_a_frag(const _Float16* row_base, int kb, int lane) {
  const int koff = (lane < 16) ? 0 : 8;
  const _Float16* p = row_base + kb + koff;
  v8h lo = *(const v8h*)(p);
  v8h hi = *(const v8h*)(p + 16);
  return combine16(lo, hi);
}

// col_base points at the first K element of this lane's N column (pre-transposed
// weights: contiguous in K); kb in {0,32,64,96}
__device__ __forceinline__ v16h load_b_frag(const _Float16* col_base, int kb, int lane) {
  const int ks = kb + ((lane < 16) ? 0 : 8);
  v8h lo = *(const v8h*)(col_base + ks);
  v8h hi = *(const v8h*)(col_base + ks + 16);
  return combine16(lo, hi);
}

// ---------------------------------------------------------------------------
// Kernel 1: per-ray view-direction embedding  ve[27] = [dir, sin(ang), cos(ang)]
// ---------------------------------------------------------------------------
__global__ void prep_rays_kernel(const float* __restrict__ rays_d,
                                 float* __restrict__ vde) {
  int r = blockIdx.x * blockDim.x + threadIdx.x;
  if (r >= N_RAYS) return;
  float dx = rays_d[r * 3 + 0];
  float dy = rays_d[r * 3 + 1];
  float dz = rays_d[r * 3 + 2];
  float inv = rsqrtf(dx * dx + dy * dy + dz * dz);
  dx *= inv; dy *= inv; dz *= inv;
  float* ve = vde + r * 27;
  ve[0] = dx; ve[1] = dy; ve[2] = dz;
  float d[3] = {dx, dy, dz};
#pragma unroll
  for (int i = 0; i < 3; ++i) {
#pragma unroll
    for (int j = 0; j < 4; ++j) {
      float ang = d[i] * (float)(1 << j);
      ve[3 + i * 4 + j]  = __sinf(ang);
      ve[15 + i * 4 + j] = __cosf(ang);
    }
  }
}

// ---------------------------------------------------------------------------
// Kernel 2: weights -> f16, transposed to [N][K] so B fragments are contiguous
//   Wt0: [128][64]  (K padded 39->64 with zeros)
//   Wt1: [128][128]
//   Wt2: [16][128]  (N padded 3->16 with zeros)
// ---------------------------------------------------------------------------
__global__ void prep_weights_kernel(const float* __restrict__ w0,
                                    const float* __restrict__ w1,
                                    const float* __restrict__ w2,
                                    _Float16* __restrict__ Wt0,
                                    _Float16* __restrict__ Wt1,
                                    _Float16* __restrict__ Wt2) {
  int t = blockIdx.x * blockDim.x + threadIdx.x;
  if (t < 128 * 64) {
    int n = t / 64, k = t % 64;
    Wt0[n * 64 + k] = (_Float16)((k < DIM0) ? w0[k * WIDTH + n] : 0.0f);
  } else if (t < 128 * 64 + 128 * 128) {
    int i = t - 128 * 64;
    int n = i / 128, k = i % 128;
    Wt1[n * 128 + k] = (_Float16)w1[k * WIDTH + n];
  } else if (t < 128 * 64 + 128 * 128 + 16 * 128) {
    int i = t - (128 * 64 + 128 * 128);
    int n = i / 128, k = i % 128;
    Wt2[n * 128 + k] = (_Float16)((n < 3) ? w2[k * 3 + n] : 0.0f);
  }
}

// ---------------------------------------------------------------------------
// Kernel 3 (fused): featurize 128 samples into LDS, then run the MLP with WMMA.
// 256 threads = 8 waves; wave w owns rows w*16 .. w*16+15 of the block tile.
// ---------------------------------------------------------------------------
__global__ __launch_bounds__(256)
void fused_mlp_kernel(const float* __restrict__ rays_o,
                      const float* __restrict__ dens_grid,
                      const float* __restrict__ k0_grid,
                      const float* __restrict__ vde,
                      const _Float16* __restrict__ Wt0,
                      const _Float16* __restrict__ Wt1,
                      const _Float16* __restrict__ Wt2,
                      const float* __restrict__ b0,
                      const float* __restrict__ b1,
                      const float* __restrict__ b2,
                      float* __restrict__ galpha,
                      float* __restrict__ grgb) {
  __shared__ _Float16 s_feat[ROWS_PER_BLOCK * KPAD];   // 16 KB
  __shared__ _Float16 s_h[ROWS_PER_BLOCK * WIDTH];     // 32 KB

  const int tid = threadIdx.x;
  const int m0  = blockIdx.x * ROWS_PER_BLOCK;

  // Warm the weight tiles (reused by every block) into cache.
  __builtin_prefetch(Wt1 + ((tid * 64) & (128 * 128 - 1)), 0, 0);

  // ---- Phase 1: featurize (threads 0..127, one sample row each) ----
  if (tid < ROWS_PER_BLOCK) {
    const int m = m0 + tid;
    const int r = m / N_SAMPLES;
    const int s = m - r * N_SAMPLES;

    const float* ve = vde + r * 27;
    const float dx = ve[0], dy = ve[1], dz = ve[2];
    const float tval = NEARP + (FARP - NEARP) * ((float)s + 0.5f) / (float)N_SAMPLES;
    float px = rays_o[r * 3 + 0] + dx * tval;
    float py = rays_o[r * 3 + 1] + dy * tval;
    float pz = rays_o[r * 3 + 2] + dz * tval;

    // contract points outside the unit cube into the background shell
    float nrm = fmaxf(fabsf(px), fmaxf(fabsf(py), fabsf(pz)));
    if (nrm > 1.0f) {
      float sc = (1.0f + BG - BG / nrm) / nrm;
      px *= sc; py *= sc; pz *= sc;
    }

    // grid coords (align_corners=True, border clamp)
    const float gscale = (float)(GRID_D - 1) / (XYZ_MAX - XYZ_MIN);
    float tx = fminf(fmaxf((px - XYZ_MIN) * gscale, 0.0f), (float)(GRID_D - 1));
    float ty = fminf(fmaxf((py - XYZ_MIN) * gscale, 0.0f), (float)(GRID_D - 1));
    float tz = fminf(fmaxf((pz - XYZ_MIN) * gscale, 0.0f), (float)(GRID_D - 1));
    int ix = min((int)tx, GRID_D - 2);
    int iy = min((int)ty, GRID_D - 2);
    int iz = min((int)tz, GRID_D - 2);
    float fx = tx - (float)ix, fy = ty - (float)iy, fz = tz - (float)iz;

    const float w000 = (1 - fx) * (1 - fy) * (1 - fz);
    const float w100 = fx * (1 - fy) * (1 - fz);
    const float w010 = (1 - fx) * fy * (1 - fz);
    const float w001 = (1 - fx) * (1 - fy) * fz;
    const float w110 = fx * fy * (1 - fz);
    const float w101 = fx * (1 - fy) * fz;
    const float w011 = (1 - fx) * fy * fz;
    const float w111 = fx * fy * fz;

    const size_t D2 = (size_t)GRID_D * GRID_D;
    const size_t D3 = D2 * GRID_D;
    const size_t o000 = (size_t)ix * D2 + (size_t)iy * GRID_D + (size_t)iz;
    const size_t o100 = o000 + D2,            o010 = o000 + GRID_D;
    const size_t o001 = o000 + 1,             o110 = o100 + GRID_D;
    const size_t o101 = o100 + 1,             o011 = o010 + 1;
    const size_t o111 = o110 + 1;

    // density -> alpha
    float dv = w000 * dens_grid[o000] + w100 * dens_grid[o100] +
               w010 * dens_grid[o010] + w001 * dens_grid[o001] +
               w110 * dens_grid[o110] + w101 * dens_grid[o101] +
               w011 * dens_grid[o011] + w111 * dens_grid[o111];
    float x = dv + ACT_SHIFT;
    float sigma = (x > 20.0f) ? x : log1pf(__expf(x));
    galpha[m] = 1.0f - __expf(-sigma * STEPDIST);

    // 12-channel k0 trilerp -> feature row (f16, K padded to 64)
    _Float16* frow = s_feat + tid * KPAD;
#pragma unroll
    for (int c = 0; c < K0_DIM; ++c) {
      const float* g = k0_grid + (size_t)c * D3;
      float v = w000 * g[o000] + w100 * g[o100] + w010 * g[o010] +
                w001 * g[o001] + w110 * g[o110] + w101 * g[o101] +
                w011 * g[o011] + w111 * g[o111];
      frow[c] = (_Float16)v;
    }
#pragma unroll
    for (int j = 0; j < 27; ++j) frow[K0_DIM + j] = (_Float16)ve[j];
#pragma unroll
    for (int k = DIM0; k < KPAD; ++k) frow[k] = (_Float16)0.0f;
  }
  __syncthreads();

  // ---- Phase 2: MLP with WMMA (all 8 waves, full EXEC) ----
  const int wv   = tid >> 5;
  const int lane = tid & 31;
  const int ln   = lane & 15;
  const int rhalf = (lane < 16) ? 0 : 8;

  // Layer 0: [16 x 64] x [64 x 128]  (2 K-steps per N-tile)
  const _Float16* my_feat = s_feat + (size_t)(wv * 16 + ln) * KPAD;
  v16h a0 = load_a_frag(my_feat, 0, lane);
  v16h a1 = load_a_frag(my_feat, 32, lane);
#pragma unroll
  for (int nt = 0; nt < 8; ++nt) {
    const _Float16* wcol = Wt0 + (size_t)(nt * 16 + ln) * KPAD;
    v8f acc = {};
    v16h bf0 = load_b_frag(wcol, 0, lane);
    acc = __builtin_amdgcn_wmma_f32_16x16x32_f16(false, a0, false, bf0, (short)0, acc, false, false);
    v16h bf1 = load_b_frag(wcol, 32, lane);
    acc = __builtin_amdgcn_wmma_f32_16x16x32_f16(false, a1, false, bf1, (short)0, acc, false, false);
    const float bias = b0[nt * 16 + ln];
    _Float16* hcol = s_h + (size_t)(wv * 16 + rhalf) * WIDTH + nt * 16 + ln;
#pragma unroll
    for (int rr = 0; rr < 8; ++rr)
      hcol[rr * WIDTH] = (_Float16)fmaxf(acc[rr] + bias, 0.0f);
  }
  __syncthreads();

  // Layer 1: [16 x 128] x [128 x 128]  (4 K-steps per N-tile)
  const _Float16* my_h = s_h + (size_t)(wv * 16 + ln) * WIDTH;
  v16h ha[4];
#pragma unroll
  for (int kb = 0; kb < 4; ++kb) ha[kb] = load_a_frag(my_h, kb * 32, lane);
  __syncthreads();   // everyone's A fragments are in registers; safe to overwrite s_h
#pragma unroll
  for (int nt = 0; nt < 8; ++nt) {
    const _Float16* wcol = Wt1 + (size_t)(nt * 16 + ln) * WIDTH;
    v8f acc = {};
#pragma unroll
    for (int kb = 0; kb < 4; ++kb) {
      v16h bf = load_b_frag(wcol, kb * 32, lane);
      acc = __builtin_amdgcn_wmma_f32_16x16x32_f16(false, ha[kb], false, bf, (short)0, acc, false, false);
    }
    const float bias = b1[nt * 16 + ln];
    _Float16* hcol = s_h + (size_t)(wv * 16 + rhalf) * WIDTH + nt * 16 + ln;
#pragma unroll
    for (int rr = 0; rr < 8; ++rr)
      hcol[rr * WIDTH] = (_Float16)fmaxf(acc[rr] + bias, 0.0f);
  }
  __syncthreads();

  // Layer 2: [16 x 128] x [128 x 16]  (cols 0..2 valid)
  v8f acc2 = {};
  const _Float16* wcol2 = Wt2 + (size_t)ln * WIDTH;
#pragma unroll
  for (int kb = 0; kb < 4; ++kb) {
    v16h ga = load_a_frag(my_h, kb * 32, lane);
    v16h bf = load_b_frag(wcol2, kb * 32, lane);
    acc2 = __builtin_amdgcn_wmma_f32_16x16x32_f16(false, ga, false, bf, (short)0, acc2, false, false);
  }
  if (ln < 3) {
    const float bias = b2[ln];
    const int rowbase = m0 + wv * 16 + rhalf;
#pragma unroll
    for (int rr = 0; rr < 8; ++rr) {
      float v = acc2[rr] + bias;
      grgb[(size_t)(rowbase + rr) * 3 + ln] = 1.0f / (1.0f + __expf(-v));
    }
  }
}

// ---------------------------------------------------------------------------
// Kernel 4: per-ray alpha compositing (serial scan over 192 samples)
// ---------------------------------------------------------------------------
__global__ void composite_kernel(const float* __restrict__ galpha,
                                 const float* __restrict__ grgb,
                                 float* __restrict__ out) {
  int r = blockIdx.x * blockDim.x + threadIdx.x;
  if (r >= N_RAYS) return;
  float T = 1.0f, wsum = 0.0f;
  float c0 = 0.0f, c1 = 0.0f, c2 = 0.0f;
  const float* ap = galpha + (size_t)r * N_SAMPLES;
  const float* cp = grgb + (size_t)r * N_SAMPLES * 3;
  for (int s = 0; s < N_SAMPLES; ++s) {
    float a = ap[s];
    float w = a * T;
    c0 += w * cp[s * 3 + 0];
    c1 += w * cp[s * 3 + 1];
    c2 += w * cp[s * 3 + 2];
    wsum += w;
    T *= (1.0f - a + 1e-10f);
  }
  float bgw = 1.0f - wsum;
  out[r * 3 + 0] = c0 + bgw;
  out[r * 3 + 1] = c1 + bgw;
  out[r * 3 + 2] = c2 + bgw;
}

// ---------------------------------------------------------------------------
extern "C" void kernel_launch(void* const* d_in, const int* in_sizes, int n_in,
                              void* d_out, int out_size, void* d_ws, size_t ws_size,
                              hipStream_t stream) {
  const float* rays_o = (const float*)d_in[0];
  const float* rays_d = (const float*)d_in[1];
  const float* dens   = (const float*)d_in[2];
  const float* k0g    = (const float*)d_in[3];
  const float* w0     = (const float*)d_in[4];
  const float* b0     = (const float*)d_in[5];
  const float* w1     = (const float*)d_in[6];
  const float* b1     = (const float*)d_in[7];
  const float* w2     = (const float*)d_in[8];
  const float* b2     = (const float*)d_in[9];
  float* out = (float*)d_out;

  // workspace layout (all offsets 256B-aligned)
  char* ws = (char*)d_ws;
  float*     vde    = (float*)(ws);                                   // 4096*27*4      = 442368
  float*     galpha = (float*)(ws + 442368);                          // 786432*4       = 3145728
  float*     grgb   = (float*)(ws + 442368 + 3145728);                // 786432*3*4     = 9437184
  _Float16*  Wt0    = (_Float16*)(ws + 442368 + 3145728 + 9437184);   // 128*64*2       = 16384
  _Float16*  Wt1    = (_Float16*)((char*)Wt0 + 16384);                // 128*128*2      = 32768
  _Float16*  Wt2    = (_Float16*)((char*)Wt1 + 32768);                // 16*128*2       = 4096

  prep_rays_kernel<<<(N_RAYS + 255) / 256, 256, 0, stream>>>(rays_d, vde);

  const int wtot = 128 * 64 + 128 * 128 + 16 * 128;
  prep_weights_kernel<<<(wtot + 255) / 256, 256, 0, stream>>>(w0, w1, w2, Wt0, Wt1, Wt2);

  fused_mlp_kernel<<<M_TOTAL / ROWS_PER_BLOCK, 256, 0, stream>>>(
      rays_o, dens, k0g, vde, Wt0, Wt1, Wt2, b0, b1, b2, galpha, grgb);

  composite_kernel<<<(N_RAYS + 255) / 256, 256, 0, stream>>>(galpha, grgb, out);
}
